// GCNEvolutionModel_83751862272247
// MI455X (gfx1250) — matrix operand order repaired and checked by reading, not compiled, not run on hardware
//
#include <hip/hip_runtime.h>
#include <hip/hip_bf16.h>

typedef __bf16 bf16;
typedef __attribute__((ext_vector_type(16))) bf16  v16bf;
typedef __attribute__((ext_vector_type(8)))  bf16  v8bf;
typedef __attribute__((ext_vector_type(4)))  bf16  v4bf;
typedef __attribute__((ext_vector_type(8)))  float v8f;

// ---------------------------------------------------------------- degrees ----
__global__ void k_zero_u32(unsigned* p, int n) {
    int i = blockIdx.x * blockDim.x + threadIdx.x;
    if (i < n) p[i] = 0u;
}

__global__ void k_deg_count(const long long* __restrict__ col, unsigned* __restrict__ deg, int E) {
    int e = blockIdx.x * blockDim.x + threadIdx.x;
    if (e < E) atomicAdd(&deg[(int)col[e]], 1u);
}

// dinv[i] = rsqrt(deg[i] + 1)   (+1 = self loop; separate buffer, deg kept for CSR)
__global__ void k_dinv(const unsigned* __restrict__ deg, float* __restrict__ dinv, int n) {
    int i = blockIdx.x * blockDim.x + threadIdx.x;
    if (i < n) dinv[i] = rsqrtf((float)(deg[i] + 1u));
}

// ------------------------------------------------------------ CSR build ------
// Exclusive scan of deg -> rowptr[0..n], single block of 1024 threads.
__global__ __launch_bounds__(1024) void k_scan_excl(const unsigned* __restrict__ deg,
                                                    unsigned* __restrict__ rowptr, int n) {
    __shared__ unsigned s[1024];
    const int t = threadIdx.x;
    const int chunk = (n + 1023) >> 10;
    const int b0 = t * chunk;
    const int b1 = (b0 + chunk < n) ? b0 + chunk : n;
    unsigned sum = 0;
    for (int i = b0; i < b1; ++i) sum += deg[i];
    s[t] = sum;
    __syncthreads();
    for (int off = 1; off < 1024; off <<= 1) {   // Hillis-Steele inclusive scan
        unsigned v = (t >= off) ? s[t - off] : 0u;
        __syncthreads();
        s[t] += v;
        __syncthreads();
    }
    unsigned run = (t == 0) ? 0u : s[t - 1];
    for (int i = b0; i < b1; ++i) { unsigned d = deg[i]; rowptr[i] = run; run += d; }
    if (t == 1023) rowptr[n] = s[1023];          // total == E
}

// One u32 atomic per edge to reserve a CSR slot; src list per destination node.
__global__ void k_csr_fill(const long long* __restrict__ row, const long long* __restrict__ col,
                           const unsigned* __restrict__ rowptr, unsigned* __restrict__ cursor,
                           unsigned* __restrict__ srcidx, int E) {
    int e = blockIdx.x * blockDim.x + threadIdx.x;
    if (e < E) {
        int c = (int)col[e];
        unsigned pos = rowptr[c] + atomicAdd(&cursor[c], 1u);
        srcidx[pos] = (unsigned)row[e];
    }
}

// ---------------------------------------------------------------- casts ------
__global__ void k_cast_bf16(const float* __restrict__ src, bf16* __restrict__ dst, long n) {
    long i = (long)blockIdx.x * blockDim.x + threadIdx.x;
    if (i < n) dst[i] = (bf16)src[i];
}

// Pack W [K x Nout, row-major f32] into the WMMA B-operand lane layout (bf16).
__global__ void k_pack_w(const float* __restrict__ W, bf16* __restrict__ Wp, int K, int Nout) {
    int tid = blockIdx.x * blockDim.x + threadIdx.x;
    if (tid >= K * Nout) return;
    int j    = tid & 15;
    int lane = (tid >> 4) & 31;
    int rest = tid >> 9;
    int kt   = rest % (K >> 5);
    int nt   = rest / (K >> 5);
    int k = kt * 32 + ((lane >> 4) << 3) + (j & 7) + ((j >> 3) << 4);
    int n = nt * 16 + (lane & 15);
    Wp[tid] = (bf16)W[k * Nout + n];
}

// ---------------------------------------------------------------- WMMA GEMM --
// C[M, NT*16] = A[M, K](bf16, row-major) * Bp(packed bf16), f32 accumulate.
template <int K, int NT>
__global__ __launch_bounds__(NT * 32) void k_gemm_wmma(const bf16* __restrict__ A,
                                                       const bf16* __restrict__ Bp,
                                                       float* __restrict__ C) {
    __shared__ __align__(16) bf16 As[16 * K];
    const int tid = threadIdx.x;
    {   // cooperative 16xK A-tile stage (rows contiguous in A)
        const uint4* src = (const uint4*)(A + (size_t)blockIdx.x * (16 * K));
        uint4* dst = (uint4*)As;
        for (int i = tid; i < (16 * K) / 8; i += NT * 32) dst[i] = src[i];
    }
    __syncthreads();

    const int lane  = tid & 31;
    const int nt    = tid >> 5;
    const int mrow  = lane & 15;
    const int khalf = (lane >> 4) << 3;
    v8f acc = {};

#pragma unroll
    for (int kt = 0; kt < K / 32; ++kt) {
        const int k0 = kt * 32 + khalf;
        v8bf lo = *(const v8bf*)(As + mrow * K + k0);
        v8bf hi = *(const v8bf*)(As + mrow * K + k0 + 16);
        v16bf a;
#pragma unroll
        for (int j = 0; j < 8; ++j) { a[j] = lo[j]; a[j + 8] = hi[j]; }
        v16bf b = *(const v16bf*)(Bp + ((size_t)(nt * (K / 32) + kt) * 32 + lane) * 16);
        acc = __builtin_amdgcn_wmma_f32_16x16x32_bf16(false, a, false, b,
                                                      (short)0, acc, false, false);
    }

    const int    NN    = NT * 16;
    const int    n     = nt * 16 + (lane & 15);
    const size_t mbase = (size_t)blockIdx.x * 16 + ((lane >> 4) << 3);
#pragma unroll
    for (int v = 0; v < 8; ++v) C[(mbase + v) * NN + n] = acc[v];
}

// ----------------------------------------------------- pull aggregation ------
// Layer 1: wave per node c. acc = dinv[c]^2*h[c] + sum_e dinv[r]dinv[c]*h[r],
// then fuse bias + relu + bf16 cast (output feeds GEMM2's A operand).  F = 128.
__global__ void k_pull_relu_bf16(const unsigned* __restrict__ rowptr, const unsigned* __restrict__ srcidx,
                                 const float* __restrict__ dinv, const float* __restrict__ h,
                                 const float* __restrict__ bias, bf16* __restrict__ out, int N) {
    const int lane = threadIdx.x & 31;
    int wid = (int)((blockIdx.x * blockDim.x + threadIdx.x) >> 5);
    const int nw = (int)((gridDim.x * blockDim.x) >> 5);
    const float4 bb = *(const float4*)(bias + lane * 4);
    for (int c = wid; c < N; c += nw) {
        const float dc = dinv[c];
        float4 hc = *(const float4*)(h + (size_t)c * 128 + lane * 4);
        float4 acc = make_float4(dc * dc * hc.x, dc * dc * hc.y, dc * dc * hc.z, dc * dc * hc.w);
        const unsigned e1 = rowptr[c + 1];
        for (unsigned e = rowptr[c]; e < e1; ++e) {
            const int r = (int)srcidx[e];
            const float w = dinv[r] * dc;
            float4 hr = *(const float4*)(h + (size_t)r * 128 + lane * 4);
            acc.x += w * hr.x; acc.y += w * hr.y; acc.z += w * hr.z; acc.w += w * hr.w;
        }
        v4bf o;
        o[0] = (bf16)fmaxf(acc.x + bb.x, 0.0f);
        o[1] = (bf16)fmaxf(acc.y + bb.y, 0.0f);
        o[2] = (bf16)fmaxf(acc.z + bb.z, 0.0f);
        o[3] = (bf16)fmaxf(acc.w + bb.w, 0.0f);
        *(v4bf*)(out + (size_t)c * 128 + lane * 4) = o;
    }
}

// Layer 2 + head: wave per node, F = 64; fuses +b2 and the 64->1 FC (+bfc)
// with a wave32 shuffle reduction. One float store per node.
__global__ void k_pull_fc(const unsigned* __restrict__ rowptr, const unsigned* __restrict__ srcidx,
                          const float* __restrict__ dinv, const float* __restrict__ h,
                          const float* __restrict__ b2, const float* __restrict__ Wfc,
                          const float* __restrict__ bfc, float* __restrict__ out, int N) {
    const int lane = threadIdx.x & 31;
    int wid = (int)((blockIdx.x * blockDim.x + threadIdx.x) >> 5);
    const int nw = (int)((gridDim.x * blockDim.x) >> 5);
    const float2 wf = *(const float2*)(Wfc + lane * 2);
    const float2 cb = *(const float2*)(b2 + lane * 2);
    for (int c = wid; c < N; c += nw) {
        const float dc = dinv[c];
        float2 hc = *(const float2*)(h + (size_t)c * 64 + lane * 2);
        float ax = dc * dc * hc.x, ay = dc * dc * hc.y;
        const unsigned e1 = rowptr[c + 1];
        for (unsigned e = rowptr[c]; e < e1; ++e) {
            const int r = (int)srcidx[e];
            const float w = dinv[r] * dc;
            float2 hr = *(const float2*)(h + (size_t)r * 64 + lane * 2);
            ax += w * hr.x; ay += w * hr.y;
        }
        float s = (ax + cb.x) * wf.x + (ay + cb.y) * wf.y;
#pragma unroll
        for (int off = 16; off > 0; off >>= 1) s += __shfl_xor(s, off, 32);
        if (lane == 0) out[c] = s + bfc[0];
    }
}

// ---------------------------------------------------------------- driver -----
extern "C" void kernel_launch(void* const* d_in, const int* in_sizes, int n_in,
                              void* d_out, int out_size, void* d_ws, size_t ws_size,
                              hipStream_t stream) {
    const float*     x   = (const float*)d_in[0];
    const long long* ei  = (const long long*)d_in[1];   // [2, E] int64
    // d_in[2] edge_weight: unused by the reference forward
    const float*     W1  = (const float*)d_in[3];
    const float*     b1  = (const float*)d_in[4];
    const float*     W2  = (const float*)d_in[5];
    const float*     b2  = (const float*)d_in[6];
    const float*     Wfc = (const float*)d_in[7];
    const float*     bfc = (const float*)d_in[8];
    float* out = (float*)d_out;

    const int IN_F = 128, HID = 128, OUT_F = 64;
    const int N = in_sizes[0] / IN_F;
    const int E = in_sizes[1] / 2;
    const long long* row = ei;
    const long long* col = ei + E;

    char* ws = (char*)d_ws;
    size_t off = 0;
    auto alloc = [&](size_t bytes) { size_t r = off; off += (bytes + 255) & ~(size_t)255; return r; };
    unsigned* deg    = (unsigned*)(ws + alloc((size_t)N * 4));        // reused as CSR cursor
    float*    dinv   = (float*)   (ws + alloc((size_t)N * 4));
    unsigned* rowptr = (unsigned*)(ws + alloc((size_t)(N + 1) * 4));
    unsigned* srcidx = (unsigned*)(ws + alloc((size_t)E * 4));
    bf16*     xbf    = (bf16*)    (ws + alloc((size_t)N * IN_F * 2)); // reused as hbf
    bf16*     w1p    = (bf16*)    (ws + alloc((size_t)IN_F * HID * 2));
    bf16*     w2p    = (bf16*)    (ws + alloc((size_t)HID * OUT_F * 2));
    float*    h1     = (float*)   (ws + alloc((size_t)N * HID * 4));  // reused as h2
    bf16*  hbf = xbf;   // xbf dead after GEMM1
    float* h2  = h1;    // h1 dead after layer-1 pull

    // 1) degree -> dinv -> CSR (one u32 atomic per edge; no float atomics anywhere)
    k_zero_u32 <<<(N + 255) / 256, 256, 0, stream>>>(deg, N);
    k_deg_count<<<(E + 255) / 256, 256, 0, stream>>>(col, deg, E);
    k_dinv     <<<(N + 255) / 256, 256, 0, stream>>>(deg, dinv, N);
    k_scan_excl<<<1, 1024, 0, stream>>>(deg, rowptr, N);
    k_zero_u32 <<<(N + 255) / 256, 256, 0, stream>>>(deg, N);         // cursor = 0
    k_csr_fill <<<(E + 255) / 256, 256, 0, stream>>>(row, col, rowptr, deg, srcidx, E);

    // 2) bf16 operand prep
    const long nx = (long)N * IN_F;
    k_cast_bf16<<<(int)((nx + 255) / 256), 256, 0, stream>>>(x, xbf, nx);
    k_pack_w<<<(IN_F * HID   + 255) / 256, 256, 0, stream>>>(W1, w1p, IN_F, HID);
    k_pack_w<<<(HID  * OUT_F + 255) / 256, 256, 0, stream>>>(W2, w2p, HID, OUT_F);

    // 3) layer 1: WMMA GEMM -> pull-aggregate (+b1, relu, bf16) -> hbf
    const int pull_grid = (N * 32 + 255) / 256;                        // wave per node
    k_gemm_wmma<128, 8><<<N / 16, 256, 0, stream>>>(xbf, w1p, h1);
    k_pull_relu_bf16<<<pull_grid, 256, 0, stream>>>(rowptr, srcidx, dinv, h1, b1, hbf, N);

    // 4) layer 2: WMMA GEMM -> pull-aggregate fused with 64->1 FC head
    k_gemm_wmma<128, 4><<<N / 16, 128, 0, stream>>>(hbf, w2p, h2);
    k_pull_fc<<<pull_grid, 256, 0, stream>>>(rowptr, srcidx, dinv, h2, b2, Wfc, bfc, out, N);
}
